// TemporalSelfAttention_18846316495048
// MI455X (gfx1250) — matrix operand order repaired and checked by reading, not compiled, not run on hardware
//
#include <hip/hip_runtime.h>
#include <cstdint>

typedef __bf16 bf16_t;
typedef bf16_t v16bf __attribute__((ext_vector_type(16)));
typedef bf16_t v8bf  __attribute__((ext_vector_type(8)));
typedef float  v8f   __attribute__((ext_vector_type(8)));
typedef unsigned int u32x4 __attribute__((ext_vector_type(4)));
typedef int          i32x4 __attribute__((ext_vector_type(4)));
typedef int          i32x8 __attribute__((ext_vector_type(8)));

#define BATCH 4
#define SEQN  2048
#define SEQS  4096
#define CDIM  256
#define C3    768

static __device__ __forceinline__ v8f wmma_bf16(v16bf a, v16bf b, v8f c) {
    return __builtin_amdgcn_wmma_f32_16x16x32_bf16(false, a, false, b, (short)0, c, false, false);
}

// Load a 16x32 A-fragment slice: two contiguous 8 x bf16 runs, 16B apart.
static __device__ __forceinline__ v16bf load_afrag(const bf16_t* p) {
    v8bf lo = *(const v8bf*)(p);
    v8bf hi = *(const v8bf*)(p + 16);
    return __builtin_shufflevector(lo, hi, 0,1,2,3,4,5,6,7,8,9,10,11,12,13,14,15);
}

// Issue a TDM flat-tile load: nelem bf16 elements from global -> LDS.
// D# packed per CDNA5 ISA ch.8 (group0: count/lds/global/type; group1: dims).
static __device__ __forceinline__ void tdm_load_tile(uint32_t lds_off,
                                                     const void* gptr,
                                                     uint32_t nelem) {
    const uint64_t ga = (uint64_t)(uintptr_t)gptr;
    u32x4 g0;
    g0[0] = 1u;                                   // count=1, user mode
    g0[1] = lds_off;                              // lds_addr
    g0[2] = (uint32_t)ga;                         // global_addr[31:0]
    g0[3] = (uint32_t)((ga >> 32) & 0x01ffffffu)  // global_addr[56:32]
          | 0x80000000u;                          // type=2 ("image")
    i32x8 g1;
    g1[0] = 0x00010000;                           // wg_mask=0, data_size=2B
    g1[1] = (int)(nelem << 16);                   // tensor_dim0[15:0]
    g1[2] = (int)((nelem >> 16) | (1u << 16));    // tensor_dim0[31:16], tensor_dim1=1
    g1[3] = (int)(nelem << 16);                   // tile_dim0 (<=65535)
    g1[4] = 1;                                    // tile_dim1=1, tile_dim2=0
    g1[5] = (int)nelem;                           // tensor_dim0_stride[31:0]
    g1[6] = 0;                                    // stride hi, dim1_stride lo
    g1[7] = 0;
    i32x4 z4 = {0, 0, 0, 0};
#if defined(__clang_major__) && (__clang_major__ >= 23)
    i32x8 z8 = {0, 0, 0, 0, 0, 0, 0, 0};
    __builtin_amdgcn_tensor_load_to_lds(g0, g1, z4, z4, z8, 0);
#else
    __builtin_amdgcn_tensor_load_to_lds(g0, g1, z4, z4, 0);
#endif
}

// ---------------------------------------------------------------------------
// Prep: transpose + convert f32 weight [K,N] -> bf16 [N,K] so B-fragments are
// two contiguous 16B runs per lane.
// ---------------------------------------------------------------------------
__global__ __launch_bounds__(256) void transpose_cvt_kernel(
    const float* __restrict__ src, bf16_t* __restrict__ dst, int K, int N)
{
    const int tid = blockIdx.x * 256 + threadIdx.x;
    if (tid >= K * N) return;
    const int k = tid / N, n = tid % N;
    dst[(size_t)n * K + k] = (bf16_t)src[tid];
}

// ---------------------------------------------------------------------------
// Kernel 1: qkv = concat(prev_x, x) @ W_qkv + b_qkv  -> q,k,v bf16 buffers
// grid (1024, 6), block 256 (8 waves).
// ---------------------------------------------------------------------------
__global__ __launch_bounds__(256) void qkv_kernel(
    const float* __restrict__ x, const float* __restrict__ prev_x,
    const bf16_t* __restrict__ Wtq,   // [768,256] bf16, transposed
    const float* __restrict__ bqkv,
    bf16_t* __restrict__ qb, bf16_t* __restrict__ kb, bf16_t* __restrict__ vb)
{
    __shared__ bf16_t As[16 * CDIM];
    const int mtile = blockIdx.x;
    const int t = threadIdx.x;

    // Stage A tile (16 rows x 256 cols of concat(prev_x, x)) as bf16 in LDS.
    {
        const int row = t >> 4;
        const int c0  = (t & 15) * 16;
        const int gm  = mtile * 16 + row;
        const int bb  = gm >> 12;
        const int s   = gm & 4095;
        const float* src = (s < SEQN)
            ? (prev_x + ((size_t)bb * SEQN + s) * CDIM)
            : (x      + ((size_t)bb * SEQN + (s - SEQN)) * CDIM);
        #pragma unroll
        for (int g = 0; g < 2; ++g) {
            v8bf tmp;
            #pragma unroll
            for (int i = 0; i < 8; ++i) tmp[i] = (bf16_t)src[c0 + g * 8 + i];
            *(v8bf*)(As + row * CDIM + c0 + g * 8) = tmp;
        }
    }
    __syncthreads();

    const int lane  = t & 31;
    const int wave  = t >> 5;
    const int ntile = blockIdx.y * 8 + wave;   // 0..47
    const int col   = lane & 15;
    const int hi    = lane >> 4;
    const int gn    = ntile * 16 + col;

    v8f acc = {};
    for (int kk = 0; kk < CDIM; kk += 32) {
        v16bf a  = load_afrag(As + col * CDIM + kk + hi * 8);
        v16bf bf = load_afrag(Wtq + (size_t)gn * CDIM + kk + hi * 8);
        acc = wmma_bf16(a, bf, acc);
    }

    const float bias = bqkv[gn];
    #pragma unroll
    for (int j = 0; j < 8; ++j) {
        const int row = j + hi * 8;
        const int gm  = mtile * 16 + row;
        const int bb  = gm >> 12;
        const int s   = gm & 4095;
        const bf16_t v = (bf16_t)(acc[j] + bias);
        const size_t base = ((size_t)bb * SEQS + s) * CDIM;
        if      (gn < CDIM)     qb[base + gn]            = v;
        else if (gn < 2 * CDIM) kb[base + gn - CDIM]     = v;
        else                    vb[base + gn - 2 * CDIM] = v;
    }
}

// ---------------------------------------------------------------------------
// Kernel 2: flash attention over S=4096 keys for the last 2048 queries/batch.
// grid (16, BATCH), block 256 (8 waves); wave handles one 16-query tile.
// K chunk staged into LDS by the Tensor Data Mover (TENSORcnt); V staged
// transposed manually (TDM cannot transpose).
// ---------------------------------------------------------------------------
__global__ __launch_bounds__(256) void attn_kernel(
    const bf16_t* __restrict__ qb, const bf16_t* __restrict__ kb,
    const bf16_t* __restrict__ vb, float* __restrict__ attn_out)
{
    __shared__ bf16_t Ks[32 * CDIM];      // [key][ch]  16 KB (TDM dest)
    __shared__ bf16_t Vst[CDIM * 32];     // [ch][key]  16 KB (transposed)
    __shared__ bf16_t Ps[8][16 * 32];     // per-wave P bounce, 8 KB

    const int bb   = blockIdx.y;
    const int t    = threadIdx.x;
    const int lane = t & 31;
    const int wave = t >> 5;
    const int col  = lane & 15;
    const int hi   = lane >> 4;
    const int qtile = blockIdx.x * 8 + wave;            // 0..127
    const size_t bbase = (size_t)bb * SEQS * CDIM;
    const int qrow = SEQN + qtile * 16 + col;           // absolute seq index

    // Preload Q A-fragments (16 queries x 256 ch bf16) with wide loads.
    v16bf qf[8];
    #pragma unroll
    for (int ks = 0; ks < 8; ++ks)
        qf[ks] = load_afrag(qb + bbase + (size_t)qrow * CDIM + ks * 32 + hi * 8);

    float m[8], l[8];
    #pragma unroll
    for (int j = 0; j < 8; ++j) { m[j] = -1e30f; l[j] = 0.f; }
    v8f acc[16];
    #pragma unroll
    for (int c = 0; c < 16; ++c) { v8f z = {}; acc[c] = z; }

    const float scale = 0.0625f;   // 1/sqrt(256)
    const uint32_t ks_lds = (uint32_t)(uintptr_t)Ks;

    for (int key0 = 0; key0 < SEQS; key0 += 32) {
        __syncthreads();
        // K chunk: async TDM DMA into LDS (one issue per block, from wave 0).
        if (wave == 0)
            tdm_load_tile(ks_lds, kb + bbase + (size_t)key0 * CDIM, 32 * CDIM);
        // V chunk: stage transposed to [ch][key] with 32-bit reads.
        {
            const uint32_t* vsrc = (const uint32_t*)(vb + bbase + (size_t)key0 * CDIM);
            #pragma unroll
            for (int i = 0; i < 16; ++i) {
                const int lin = t + i * 256;        // one ch-pair of one key
                const int key = lin >> 7;           // 128 u32 per key row
                const int c0  = (lin & 127) * 2;
                const uint32_t w = vsrc[lin];
                Vst[(c0    ) * 32 + key] = __builtin_bit_cast(bf16_t, (uint16_t)(w & 0xffffu));
                Vst[(c0 + 1) * 32 + key] = __builtin_bit_cast(bf16_t, (uint16_t)(w >> 16));
            }
        }
        // Prefetch next V chunk while this one is consumed.
        if (key0 + 32 < SEQS)
            __builtin_prefetch(vb + bbase + (size_t)(key0 + 32) * CDIM + t * 32, 0, 1);
        if (wave == 0)
            __builtin_amdgcn_s_wait_tensorcnt(0);
        __syncthreads();

        // Scores: S = Q @ K^T (two 16x16 tiles covering 32 keys).
        v8f s0 = {}, s1 = {};
        #pragma unroll
        for (int ks = 0; ks < 8; ++ks) {
            v16bf b0 = load_afrag(Ks + col        * CDIM + ks * 32 + hi * 8);
            v16bf b1 = load_afrag(Ks + (col + 16) * CDIM + ks * 32 + hi * 8);
            s0 = wmma_bf16(qf[ks], b0, s0);
            s1 = wmma_bf16(qf[ks], b1, s1);
        }

        // Online softmax. Row r = j + hi*8 lives across one 16-lane half.
        float alpha[8];
        #pragma unroll
        for (int j = 0; j < 8; ++j) {
            float a0 = s0[j] * scale;
            float a1 = s1[j] * scale;
            float mx = fmaxf(a0, a1);
            #pragma unroll
            for (int off = 1; off < 16; off <<= 1)
                mx = fmaxf(mx, __shfl_xor(mx, off, 32));
            const float mn = fmaxf(m[j], mx);
            const float p0 = __expf(a0 - mn);
            const float p1 = __expf(a1 - mn);
            s0[j] = p0; s1[j] = p1;
            float sum = p0 + p1;
            #pragma unroll
            for (int off = 1; off < 16; off <<= 1)
                sum += __shfl_xor(sum, off, 32);
            alpha[j] = __expf(m[j] - mn);
            l[j] = l[j] * alpha[j] + sum;
            m[j] = mn;
        }

        // P: C-layout -> A-layout via per-wave LDS bounce (bf16).
        bf16_t* Pw = Ps[wave];
        #pragma unroll
        for (int j = 0; j < 8; ++j) {
            const int row = j + hi * 8;
            Pw[row * 32 + col]      = (bf16_t)s0[j];
            Pw[row * 32 + col + 16] = (bf16_t)s1[j];
        }
        v16bf pf = load_afrag(Pw + col * 32 + hi * 8);

        // acc = acc*alpha + P @ V  (16 channel tiles, K=32 per wmma).
        #pragma unroll
        for (int c = 0; c < 16; ++c) {
            #pragma unroll
            for (int j = 0; j < 8; ++j) acc[c][j] *= alpha[j];
            v16bf vf = load_afrag(Vst + (c * 16 + col) * 32 + hi * 8);
            acc[c] = wmma_bf16(pf, vf, acc[c]);
        }
    }

    // Normalize and store f32 attention output [B, 2048, 256].
    const size_t obase = ((size_t)bb * SEQN + qtile * 16) * CDIM;
    #pragma unroll
    for (int c = 0; c < 16; ++c) {
        #pragma unroll
        for (int j = 0; j < 8; ++j) {
            const int row = j + hi * 8;
            attn_out[obase + (size_t)row * CDIM + c * 16 + col] = acc[c][j] / l[j];
        }
    }
}

// ---------------------------------------------------------------------------
// Kernel 3: out = attn_out @ W_out + b_out  -> d_out f32 [4,2048,256]
// grid (512, 2), block 256 (8 waves).
// ---------------------------------------------------------------------------
__global__ __launch_bounds__(256) void outproj_kernel(
    const float* __restrict__ attn_out,
    const bf16_t* __restrict__ Wto,   // [256,256] bf16, transposed
    const float* __restrict__ bout, float* __restrict__ out)
{
    __shared__ bf16_t As[16 * CDIM];
    const int mtile = blockIdx.x;      // 0..511
    const int t = threadIdx.x;
    {
        const int row = t >> 4;
        const int c0  = (t & 15) * 16;
        const float* src = attn_out + ((size_t)(mtile * 16 + row)) * CDIM;
        #pragma unroll
        for (int g = 0; g < 2; ++g) {
            v8bf tmp;
            #pragma unroll
            for (int i = 0; i < 8; ++i) tmp[i] = (bf16_t)src[c0 + g * 8 + i];
            *(v8bf*)(As + row * CDIM + c0 + g * 8) = tmp;
        }
    }
    __syncthreads();

    const int lane  = t & 31;
    const int wave  = t >> 5;
    const int ntile = blockIdx.y * 8 + wave;  // 0..15
    const int col   = lane & 15;
    const int hi    = lane >> 4;
    const int gn    = ntile * 16 + col;

    v8f acc = {};
    for (int kk = 0; kk < CDIM; kk += 32) {
        v16bf a  = load_afrag(As + col * CDIM + kk + hi * 8);
        v16bf bf = load_afrag(Wto + (size_t)gn * CDIM + kk + hi * 8);
        acc = wmma_bf16(a, bf, acc);
    }

    const float bias = bout[gn];
    #pragma unroll
    for (int j = 0; j < 8; ++j) {
        const size_t gm = (size_t)mtile * 16 + j + hi * 8;
        out[gm * CDIM + gn] = acc[j] + bias;
    }
}

// ---------------------------------------------------------------------------
extern "C" void kernel_launch(void* const* d_in, const int* in_sizes, int n_in,
                              void* d_out, int out_size, void* d_ws, size_t ws_size,
                              hipStream_t stream) {
    const float* x      = (const float*)d_in[0];
    const float* prev_x = (const float*)d_in[1];
    const float* Wqkv   = (const float*)d_in[2];
    const float* bqkv   = (const float*)d_in[3];
    const float* Wout   = (const float*)d_in[4];
    const float* bout   = (const float*)d_in[5];
    float* out = (float*)d_out;

    // Workspace layout:
    //   q/k/v bf16 (8 MiB each) + attn_out f32 (8 MiB) + bf16 weights.
    char* ws = (char*)d_ws;
    const size_t qkv_bytes = (size_t)BATCH * SEQS * CDIM * sizeof(bf16_t); // 8 MiB
    bf16_t* qb  = (bf16_t*)(ws);
    bf16_t* kb  = (bf16_t*)(ws + qkv_bytes);
    bf16_t* vb  = (bf16_t*)(ws + 2 * qkv_bytes);
    float*  ao  = (float*) (ws + 3 * qkv_bytes);
    const size_t ao_bytes = (size_t)BATCH * SEQN * CDIM * sizeof(float);   // 8 MiB
    bf16_t* wtq = (bf16_t*)(ws + 3 * qkv_bytes + ao_bytes);
    bf16_t* wto = (bf16_t*)(ws + 3 * qkv_bytes + ao_bytes + (size_t)C3 * CDIM * sizeof(bf16_t));

    transpose_cvt_kernel<<<(CDIM * C3 + 255) / 256, 256, 0, stream>>>(Wqkv, wtq, CDIM, C3);
    transpose_cvt_kernel<<<(CDIM * CDIM + 255) / 256, 256, 0, stream>>>(Wout, wto, CDIM, CDIM);

    qkv_kernel<<<dim3((BATCH * SEQS) / 16, C3 / (16 * 8)), 256, 0, stream>>>(
        x, prev_x, wtq, bqkv, qb, kb, vb);
    attn_kernel<<<dim3((SEQN / 16) / 8, BATCH), 256, 0, stream>>>(
        qb, kb, vb, ao);
    outproj_kernel<<<dim3((BATCH * SEQN) / 16, CDIM / (16 * 8)), 256, 0, stream>>>(
        ao, wto, bout, out);
}
